// MultiHeadAttention_73418170958474
// MI455X (gfx1250) — matrix operand order repaired and checked by reading, not compiled, not run on hardware
//
#include <hip/hip_runtime.h>
#include <hip/hip_bf16.h>

// ---------------------------------------------------------------------------
// MHA forward for (B,N,H,D) = (16,512,8,128) on gfx1250 (CDNA5, wave32).
// All GEMMs run on v_wmma_f32_16x16x32_bf16 (bf16 operands, f32 accum).
// Workspace usage: ~67 MB (bf16 copies of x/weights + Q/K/V^T/Y staging).
// ---------------------------------------------------------------------------

#define B_  16
#define N_  512
#define H_  8
#define D_  128
#define HD_ 1024

typedef __attribute__((ext_vector_type(16))) __bf16 v16bf;
typedef __attribute__((ext_vector_type(8)))  float  v8f;

union FragAB { v16bf v; uint4 u[2]; };

__device__ __forceinline__ unsigned short f2bf(float f) {
    union { float f; unsigned int u; } a; a.f = f;
    unsigned int r = a.u + 0x7fffu + ((a.u >> 16) & 1u);   // round-to-nearest-even
    return (unsigned short)(r >> 16);
}

// ---- fragment pointers (computed once per tile, then immediate offsets) ----
// A-fragment (16x32 bf16): lane holds row (lane&15); chunks at k0+8*(lane>>4)
// and +16 more (ISA 7.12.2). Pointer bakes in row and lane phase.
__device__ __forceinline__ const unsigned short*
a_ptr(const unsigned short* base, int ld, int row0, int lane) {
    return base + (row0 + (lane & 15)) * ld + ((lane >> 4) << 3);
}
__device__ __forceinline__ v16bf load_a_at(const unsigned short* __restrict__ p,
                                           int k0) {
    FragAB f;
    f.u[0] = *reinterpret_cast<const uint4*>(p + k0);
    f.u[1] = *reinterpret_cast<const uint4*>(p + k0 + 16);
    return f.v;
}
// B-fragment (32x16 bf16): lane holds column (lane&15); chunks at k0+8*(lane>>4)
// and +16 more (same K order as A). Source must be K-contiguous per column.
__device__ __forceinline__ const unsigned short*
b_ptr(const unsigned short* base, int ld, int col0, int lane) {
    return base + (col0 + (lane & 15)) * ld + ((lane >> 4) << 3);
}
__device__ __forceinline__ v16bf load_b_at(const unsigned short* __restrict__ p,
                                           int k0) {
    FragAB f;
    f.u[0] = *reinterpret_cast<const uint4*>(p + k0);
    f.u[1] = *reinterpret_cast<const uint4*>(p + k0 + 16);
    return f.v;
}

__device__ __forceinline__ v8f wmma_bf16(v16bf a, v16bf b, v8f c) {
    return __builtin_amdgcn_wmma_f32_16x16x32_bf16(false, a, false, b,
                                                   (short)0, c, false, false);
}

// ------------------------------ prep kernels -------------------------------

__global__ void cvt_x_kernel(const float* __restrict__ x,
                             unsigned short* __restrict__ xb, int n) {
    int i = blockIdx.x * 256 + threadIdx.x;
    if (i < n) xb[i] = f2bf(x[i]);
}

// in: [R][C] f32 row-major  ->  out: [C][R] bf16 (K-contiguous per column)
__global__ void tcvt_kernel(const float* __restrict__ in,
                            unsigned short* __restrict__ out, int R, int C) {
    int i = blockIdx.x * 256 + threadIdx.x;
    if (i < R * C) {
        int r = i / C, c = i % C;
        out[(size_t)c * R + r] = f2bf(in[i]);
    }
}

// --------------------------- QKV projection --------------------------------
// One projection (P = 0:Q, 1:K, 2:V) of a 16x64 tile; compile-time
// specialized so the store path is branch-free straight-line code.
template <int P>
__device__ __forceinline__ void proj_store(const unsigned short* __restrict__ WT,
                                           const float* __restrict__ bias,
                                           const float* __restrict__ mrow,
                                           v16bf a0, v16bf a1, v16bf a2, v16bf a3,
                                           int b, int mtile, int nbase, int lane,
                                           unsigned short* __restrict__ dst) {
    const float scale = 0.08838834764831845f;   // D^-0.5
    const int rbase = mtile + ((lane >> 4) << 3);

    #pragma unroll
    for (int ns = 0; ns < 4; ++ns) {
        const int col0 = nbase + ns * 16;
        const unsigned short* bp = b_ptr(WT, D_, col0, lane);
        v8f acc = {};
        acc = wmma_bf16(a0, load_b_at(bp,  0), acc);
        acc = wmma_bf16(a1, load_b_at(bp, 32), acc);
        acc = wmma_bf16(a2, load_b_at(bp, 64), acc);
        acc = wmma_bf16(a3, load_b_at(bp, 96), acc);

        const int c = col0 + (lane & 15);
        const int h = c >> 7;
        const int d = c & 127;
        const float bsv = bias[c];
        if (P == 2) {
            unsigned short* vp =
                dst + ((size_t)(b * H_ + h) * D_ + d) * N_ + rbase;
            #pragma unroll
            for (int r = 0; r < 8; ++r)
                vp[r] = f2bf((acc[r] + bsv) * mrow[r]);
        } else {
            unsigned short* qp =
                dst + ((size_t)(b * H_ + h) * N_ + rbase) * D_ + d;
            #pragma unroll
            for (int r = 0; r < 8; ++r) {
                float v = (acc[r] + bsv) * mrow[r];
                qp[r * D_] = f2bf(P == 0 ? v * scale : v);
            }
        }
    }
}

// grid (4, 32, 16), block 128 (4 waves). Wave computes a 16x64 tile of the
// [512x128]x[128x1024] projection for one batch, for Q, K and V (A reused).
__global__ __launch_bounds__(128)
void qkv_proj_kernel(const unsigned short* __restrict__ xb,
                     const unsigned short* __restrict__ WqT,
                     const unsigned short* __restrict__ WkT,
                     const unsigned short* __restrict__ WvT,
                     const float* __restrict__ bq, const float* __restrict__ bk,
                     const float* __restrict__ bv, const float* __restrict__ mask,
                     unsigned short* __restrict__ q,
                     unsigned short* __restrict__ k,
                     unsigned short* __restrict__ vT) {
    const int lane  = threadIdx.x & 31;
    const int wave  = threadIdx.x >> 5;
    const int b     = blockIdx.z;
    const int mtile = blockIdx.y * 16;
    const int nbase = (blockIdx.x * 4 + wave) * 64;

    const unsigned short* ap = a_ptr(xb + (size_t)b * N_ * D_, D_, mtile, lane);
    v16bf a0 = load_a_at(ap,  0);
    v16bf a1 = load_a_at(ap, 32);
    v16bf a2 = load_a_at(ap, 64);
    v16bf a3 = load_a_at(ap, 96);

    const int rbase = mtile + ((lane >> 4) << 3);
    float mrow[8];
    const float* mp = mask + b * N_ + rbase;
    #pragma unroll
    for (int r = 0; r < 8; ++r) mrow[r] = mp[r];

    proj_store<0>(WqT, bq, mrow, a0, a1, a2, a3, b, mtile, nbase, lane, q);
    proj_store<1>(WkT, bk, mrow, a0, a1, a2, a3, b, mtile, nbase, lane, k);
    proj_store<2>(WvT, bv, mrow, a0, a1, a2, a3, b, mtile, nbase, lane, vT);
}

// ------------------------------ attention ----------------------------------
// grid (32, 8, 16), block 128 (4 waves). Workgroup owns 16 query rows of one
// (b,h): scores -> LDS, cooperative softmax, P(bf16) -> LDS, P@V via WMMA
// with LDS A-fragments, partials merged with ds_add_f32.
__global__ __launch_bounds__(128)
void attn_kernel(const unsigned short* __restrict__ q,
                 const unsigned short* __restrict__ k,
                 const unsigned short* __restrict__ vT,
                 const float* __restrict__ dist, const float* __restrict__ mask,
                 unsigned short* __restrict__ y) {
    __shared__ float          sS[16][512];   // scores / exp values (f32)
    __shared__ unsigned short sP[16][512];   // probabilities (bf16)
    __shared__ float          sY[16][128];   // output accumulator
    __shared__ float          sRed[16][8];   // row reductions
    __shared__ float          sCM[512];      // mask row for this batch

    const int tid  = threadIdx.x;
    const int lane = tid & 31;
    const int wave = tid >> 5;
    const int mtile = blockIdx.x * 16;
    const int h     = blockIdx.y;
    const int b     = blockIdx.z;

    const unsigned short* qh = q  + ((size_t)(b * H_ + h)) * N_ * D_;
    const unsigned short* kh = k  + ((size_t)(b * H_ + h)) * N_ * D_;
    const unsigned short* vh = vT + ((size_t)(b * H_ + h)) * D_ * N_;

    // warm caches for this wave's K-slab / V-slab (global_prefetch_b8)
    #pragma unroll 4
    for (int i = lane; i < 128; i += 32) {
        __builtin_prefetch(kh + (size_t)(wave * 128 + i) * D_, 0, 2);
        __builtin_prefetch(vh + (size_t)i * N_ + wave * 128, 0, 2);
    }

    for (int i = tid; i < 512; i += 128) sCM[i] = mask[b * N_ + i];
    for (int i = tid; i < 2048; i += 128) (&sY[0][0])[i] = 0.0f;
    __syncthreads();

    // ---- phase 1: S = Q K^T + dist, masked (wave w -> cols [128w,128w+128))
    const unsigned short* ap = a_ptr(qh, D_, mtile, lane);
    v16bf a0 = load_a_at(ap,  0);
    v16bf a1 = load_a_at(ap, 32);
    v16bf a2 = load_a_at(ap, 64);
    v16bf a3 = load_a_at(ap, 96);

    const int lrbase = (lane >> 4) << 3;        // local row base for C frags
    #pragma unroll
    for (int cs = 0; cs < 8; ++cs) {
        const int col0 = wave * 128 + cs * 16;
        const unsigned short* bp = b_ptr(kh, D_, col0, lane);
        v8f acc = {};
        acc = wmma_bf16(a0, load_b_at(bp,  0), acc);
        acc = wmma_bf16(a1, load_b_at(bp, 32), acc);
        acc = wmma_bf16(a2, load_b_at(bp, 64), acc);
        acc = wmma_bf16(a3, load_b_at(bp, 96), acc);

        const int col = col0 + (lane & 15);
        const float cm = sCM[col];
        const float* dp = dist + ((size_t)b * N_ + mtile + lrbase) * N_ + col;
        float* sp = &sS[lrbase][col];
        #pragma unroll
        for (int r = 0; r < 8; ++r) {
            const float am = cm * sCM[mtile + lrbase + r];
            const float s  = acc[r] + dp[r * N_];
            sp[r * 512] = (am == 0.0f) ? -1.0e9f : s;
        }
    }
    __syncthreads();

    // ---- phase 2: row softmax over 512 cols (8 threads per row)
    const int rrow = tid >> 3;
    const int sub  = tid & 7;
    {
        float lmax = -3.4e38f;
        for (int c = sub; c < 512; c += 8) lmax = fmaxf(lmax, sS[rrow][c]);
        sRed[rrow][sub] = lmax;
    }
    __syncthreads();
    float rmax = -3.4e38f;
    #pragma unroll
    for (int i = 0; i < 8; ++i) rmax = fmaxf(rmax, sRed[rrow][i]);
    float lsum = 0.0f;
    for (int c = sub; c < 512; c += 8) {
        const float e = __expf(sS[rrow][c] - rmax);
        sS[rrow][c] = e;
        lsum += e;
    }
    __syncthreads();
    sRed[rrow][sub] = lsum;
    __syncthreads();
    float rsum = 0.0f;
    #pragma unroll
    for (int i = 0; i < 8; ++i) rsum += sRed[rrow][i];
    const float inv = 1.0f / rsum;
    const float rm  = sCM[mtile + rrow];
    for (int c = sub; c < 512; c += 8) {
        const float p = (rm == 0.0f || sCM[c] == 0.0f) ? 0.0f : sS[rrow][c] * inv;
        sP[rrow][c] = f2bf(p);
    }
    __syncthreads();

    // ---- phase 3: Y += P V  (wave w -> K-slice [128w,128w+128))
    const unsigned short* pp = &sP[0][0] + (lane & 15) * 512 + ((lane >> 4) << 3);
    const int kbase = wave * 128;
    #pragma unroll
    for (int dsb = 0; dsb < 8; ++dsb) {
        const int d0 = dsb * 16;
        const unsigned short* bp = b_ptr(vh, N_, d0, lane);
        v8f acc = {};
        acc = wmma_bf16(load_a_at(pp, kbase +  0), load_b_at(bp, kbase +  0), acc);
        acc = wmma_bf16(load_a_at(pp, kbase + 32), load_b_at(bp, kbase + 32), acc);
        acc = wmma_bf16(load_a_at(pp, kbase + 64), load_b_at(bp, kbase + 64), acc);
        acc = wmma_bf16(load_a_at(pp, kbase + 96), load_b_at(bp, kbase + 96), acc);

        const int d = d0 + (lane & 15);
        float* yp = &sY[lrbase][d];
        #pragma unroll
        for (int r = 0; r < 8; ++r)
            atomicAdd(yp + r * 128, acc[r]);                  // ds_add_f32
    }
    __syncthreads();

    // ---- phase 4: store Y (bf16) to [B][N][H*D]
    for (int i = tid; i < 2048; i += 128) {
        const int m = i >> 7, d = i & 127;
        y[((size_t)(b * N_ + mtile + m)) * HD_ + h * D_ + d] = f2bf(sY[m][d]);
    }
}

// --------------------------- output projection -----------------------------
// grid (512), block 128 (4 waves). Wave computes a 16x32 tile of
// [8192x1024] x [1024x128]; bias + row mask applied on the f32 store.
__global__ __launch_bounds__(128)
void out_proj_kernel(const unsigned short* __restrict__ y,
                     const unsigned short* __restrict__ WoT,
                     const float* __restrict__ bo, const float* __restrict__ mask,
                     float* __restrict__ out) {
    const int lane  = threadIdx.x & 31;
    const int wave  = threadIdx.x >> 5;
    const int mtile = blockIdx.x * 16;
    const int c0    = wave * 32;

    const unsigned short* ap  = a_ptr(y, HD_, mtile, lane);
    const unsigned short* bpA = b_ptr(WoT, HD_, c0, lane);
    const unsigned short* bpB = b_ptr(WoT, HD_, c0 + 16, lane);

    v8f acc0 = {}, acc1 = {};
    #pragma unroll 8
    for (int kk = 0; kk < 32; ++kk) {
        const int k0 = kk * 32;
        v16bf a = load_a_at(ap, k0);
        acc0 = wmma_bf16(a, load_b_at(bpA, k0), acc0);
        acc1 = wmma_bf16(a, load_b_at(bpB, k0), acc1);
    }

    const int rbase = mtile + ((lane >> 4) << 3);
    const int cA = c0 + (lane & 15);
    const float bA = bo[cA], bB = bo[cA + 16];
    float* op = out + (size_t)rbase * D_ + cA;
    #pragma unroll
    for (int r = 0; r < 8; ++r) {
        const int row = rbase + r;
        const float mv = mask[(row >> 9) * N_ + (row & 511)];
        op[r * D_]      = (acc0[r] + bA) * mv;
        op[r * D_ + 16] = (acc1[r] + bB) * mv;
    }
}

// ------------------------------- launcher ----------------------------------

extern "C" void kernel_launch(void* const* d_in, const int* in_sizes, int n_in,
                              void* d_out, int out_size, void* d_ws, size_t ws_size,
                              hipStream_t stream) {
    const float* x    = (const float*)d_in[0];   // [16][512][128]
    const float* dist = (const float*)d_in[1];   // [16][512][512]
    const float* mask = (const float*)d_in[2];   // [16][512]
    const float* Wq   = (const float*)d_in[3];   // [128][1024]
    const float* bq   = (const float*)d_in[4];
    const float* Wk   = (const float*)d_in[5];
    const float* bk   = (const float*)d_in[6];
    const float* Wv   = (const float*)d_in[7];
    const float* bv   = (const float*)d_in[8];
    const float* Wo   = (const float*)d_in[9];   // [1024][128]
    const float* bo   = (const float*)d_in[10];
    float* outp       = (float*)d_out;           // [16][512][128]

    size_t off = 0;
    auto carve = [&](size_t bytes) -> void* {
        void* p = (char*)d_ws + off;
        off = (off + bytes + 255) & ~(size_t)255;
        return p;
    };
    unsigned short* xb  = (unsigned short*)carve((size_t)B_ * N_ * D_ * 2);       //  2 MB
    unsigned short* WqT = (unsigned short*)carve((size_t)D_ * HD_ * 2);           // 256 KB
    unsigned short* WkT = (unsigned short*)carve((size_t)D_ * HD_ * 2);
    unsigned short* WvT = (unsigned short*)carve((size_t)D_ * HD_ * 2);
    unsigned short* WoT = (unsigned short*)carve((size_t)D_ * HD_ * 2);
    unsigned short* qw  = (unsigned short*)carve((size_t)B_ * H_ * N_ * D_ * 2);  // 16 MB
    unsigned short* kw  = (unsigned short*)carve((size_t)B_ * H_ * N_ * D_ * 2);  // 16 MB
    unsigned short* vTw = (unsigned short*)carve((size_t)B_ * H_ * N_ * D_ * 2);  // 16 MB
    unsigned short* yw  = (unsigned short*)carve((size_t)B_ * N_ * HD_ * 2);      // 16 MB
    (void)ws_size; (void)n_in; (void)in_sizes; (void)out_size;

    cvt_x_kernel<<<(B_ * N_ * D_ + 255) / 256, 256, 0, stream>>>(x, xb, B_ * N_ * D_);
    tcvt_kernel<<<(D_ * HD_ + 255) / 256, 256, 0, stream>>>(Wq, WqT, D_, HD_);
    tcvt_kernel<<<(D_ * HD_ + 255) / 256, 256, 0, stream>>>(Wk, WkT, D_, HD_);
    tcvt_kernel<<<(D_ * HD_ + 255) / 256, 256, 0, stream>>>(Wv, WvT, D_, HD_);
    tcvt_kernel<<<(HD_ * D_ + 255) / 256, 256, 0, stream>>>(Wo, WoT, HD_, D_);

    qkv_proj_kernel<<<dim3(4, 32, B_), 128, 0, stream>>>(xb, WqT, WkT, WvT,
                                                         bq, bk, bv, mask,
                                                         qw, kw, vTw);
    attn_kernel<<<dim3(N_ / 16, H_, B_), 128, 0, stream>>>(qw, kw, vTw, dist, mask, yw);

    out_proj_kernel<<<dim3((B_ * N_) / 16), 128, 0, stream>>>(yw, WoT, bo, mask, outp);
}
